// GroupedMultiQueryAttention_70351564308899
// MI455X (gfx1250) — matrix operand order repaired and checked by reading, not compiled, not run on hardware
//
#include <hip/hip_runtime.h>

// ---------------------------------------------------------------------------
// GQA attention block for MI455X (gfx1250, wave32, WMMA).
// bf16 WMMA math, f32 accumulate. GEMM tiles staged by the Tensor Data Mover
// (tensor_load_to_lds, TENSORcnt); attention tiles staged by per-lane async
// loads (global_load_async_to_lds_b128, ASYNCcnt). Double-buffered LDS.
// ---------------------------------------------------------------------------

typedef unsigned short u16;
typedef __bf16 bf16;
typedef bf16  v16bf __attribute__((ext_vector_type(16)));
typedef float v8f   __attribute__((ext_vector_type(8)));
typedef unsigned v4u __attribute__((ext_vector_type(4)));
typedef int      v8i __attribute__((ext_vector_type(8)));
typedef int      v4i __attribute__((ext_vector_type(4)));

union Frag {
    uint4 u[2];   // two 16B chunks = 16 bf16
    v16bf v;
};

__device__ __forceinline__ u16 f2bf(float f) {
    unsigned u = __float_as_uint(f);
    unsigned r = u + 0x7FFFu + ((u >> 16) & 1u);   // round-to-nearest-even
    return (u16)(r >> 16);
}

// async copy of 16 bytes global -> LDS (per lane), tracked by ASYNCcnt.
// LDS generic pointer: aperture is 4GB-aligned, low 32 bits == LDS byte offset.
__device__ __forceinline__ void async_b128(const u16* g, u16* l) {
    unsigned lds = (unsigned)(unsigned long long)l;
    asm volatile("global_load_async_to_lds_b128 %0, %1, off"
                 :: "v"(lds), "v"(g) : "memory");
}

// Tensor Data Mover: DMA a 2D tile (tile_w elems x tile_h rows, 2B elems) from
// a row-major tensor (row stride = stride_elems) into LDS at lds_off, adding
// pad_amount DWORDs of LDS padding every (2<<pad_int_code) DWORDs.
// D# layout per CDNA5 ISA 8.3/8.4; groups 2/3 zero (2D tensor).
// This toolchain's builtin takes 6 args (extra int32x8, zero-filled).
__device__ __forceinline__ void tdm_load_2d(unsigned lds_off, const void* gaddr,
                                            unsigned tile_w, unsigned tile_h,
                                            unsigned tensor_w, unsigned tensor_h,
                                            unsigned stride_elems,
                                            unsigned pad_int_code,
                                            unsigned pad_amt_code) {
    unsigned long long ga = (unsigned long long)gaddr;
    v4u g0;
    g0[0] = 1u;                                            // count=1 (valid user D#)
    g0[1] = lds_off;                                       // lds_addr (bytes)
    g0[2] = (unsigned)(ga & 0xFFFFFFFFu);                  // global_addr[31:0]
    g0[3] = (unsigned)((ga >> 32) & 0x01FFFFFFu) | (2u << 30); // [56:32] | type=2
    v8i g1;
    g1[0] = (int)((1u << 16)                               // data_size = 2B
                  | (1u << 20)                             // pad_enable
                  | (pad_int_code << 22) | (pad_amt_code << 25));
    g1[1] = (int)((tensor_w & 0xFFFFu) << 16);             // tensor_dim0 lo
    g1[2] = (int)(((tensor_w >> 16) & 0xFFFFu)             // tensor_dim0 hi
                  | ((tensor_h & 0xFFFFu) << 16));         // tensor_dim1 lo
    g1[3] = (int)(((tensor_h >> 16) & 0xFFFFu)             // tensor_dim1 hi
                  | (tile_w << 16));                       // tile_dim0
    g1[4] = (int)tile_h;                                   // tile_dim1 (tile_dim2=0)
    g1[5] = (int)stride_elems;                             // tensor_dim0_stride lo
    g1[6] = 0;
    g1[7] = 0;
    v4i z4 = {0, 0, 0, 0};
    v8i z8 = {0, 0, 0, 0, 0, 0, 0, 0};
    __builtin_amdgcn_tensor_load_to_lds(g0, g1, z4, z4, z8, 0);
}

// ---------------------------------------------------------------------------
// fp32 -> bf16 elementwise (for x)
// ---------------------------------------------------------------------------
__global__ __launch_bounds__(256) void cvt_bf16(const float4* __restrict__ in,
                                                u16* __restrict__ out, int n4) {
    int i = blockIdx.x * 256 + threadIdx.x;
    if (i < n4) {
        float4 f = in[i];
        unsigned lo = (unsigned)f2bf(f.x) | ((unsigned)f2bf(f.y) << 16);
        unsigned hi = (unsigned)f2bf(f.z) | ((unsigned)f2bf(f.w) << 16);
        *(uint2*)&out[(size_t)i * 4] = make_uint2(lo, hi);
    }
}

// ---------------------------------------------------------------------------
// fp32 [K][N] -> bf16 [N][K]
// ---------------------------------------------------------------------------
__global__ __launch_bounds__(256) void transpose_cvt(const float* __restrict__ in,
                                                     u16* __restrict__ out,
                                                     int K, int N) {
    __shared__ u16 t[32][36];
    const int n0 = blockIdx.x * 32, k0 = blockIdx.y * 32;
    const int tid = threadIdx.x;
    {
        int r = tid >> 3, c = (tid & 7) * 4;
        float4 f = *(const float4*)&in[(size_t)(k0 + r) * N + n0 + c];
        t[r][c + 0] = f2bf(f.x);
        t[r][c + 1] = f2bf(f.y);
        t[r][c + 2] = f2bf(f.z);
        t[r][c + 3] = f2bf(f.w);
    }
    __syncthreads();
    {
        int rn = tid >> 3, ck = (tid & 7) * 4;
        unsigned lo = (unsigned)t[ck + 0][rn] | ((unsigned)t[ck + 1][rn] << 16);
        unsigned hi = (unsigned)t[ck + 2][rn] | ((unsigned)t[ck + 3][rn] << 16);
        *(uint2*)&out[(size_t)(n0 + rn) * K + k0 + ck] = make_uint2(lo, hi);
    }
}

// ---------------------------------------------------------------------------
// bf16 GEMM: C[M][N] = A[M][K] * Bt[N][K]^T
// 128(M) x 256(N) tile / block, 8 waves in 2x4, wave = 64x64 = 4x4 WMMA,
// BK = 32. Tiles staged by TDM (wave 0 issues 2 descriptors per stage),
// double-buffered; LDS rows = 64B data + 16B pad via TDM pad fields.
// ---------------------------------------------------------------------------
template <bool OUT_F32>
__global__ __launch_bounds__(256) void gemm_bf16(const u16* __restrict__ A,
                                                 const u16* __restrict__ Bt,
                                                 void* __restrict__ C,
                                                 int M, int N, int K) {
    __shared__ u16 As[2][128][40];   // 32 data + 8 pad (row 80B)
    __shared__ u16 Bs[2][256][40];

    const int tid  = threadIdx.x;
    const int wave = tid >> 5;
    const int l16  = tid & 15;
    const int hl   = (tid >> 4) & 1;
    const int wm   = (tid >> 5) & 1;   // 0..1  (64 rows)
    const int wn   = tid >> 6;         // 0..3  (64 cols)
    const int m0   = blockIdx.y * 128;
    const int n0   = blockIdx.x * 256;

    v8f acc[4][4] = {};
    const int nst = K >> 5;

    // pad codes: interval 16 DWORDs (code 3) = 64B of data, amount 4 DWORDs
    // (code 3) = 16B pad -> 80B LDS row stride.
    if (wave == 0) {
        tdm_load_2d((unsigned)(unsigned long long)&As[0][0][0],
                    &A[(size_t)m0 * K], 32, 128, K, M, K, 3, 3);
        tdm_load_2d((unsigned)(unsigned long long)&Bs[0][0][0],
                    &Bt[(size_t)n0 * K], 32, 256, K, N, K, 3, 3);
    }

    for (int k = 0; k < nst; ++k) {
        const int cur = k & 1;
        if (wave == 0) {
            if (k + 1 < nst) {
                const int k1 = (k + 1) << 5;
                tdm_load_2d((unsigned)(unsigned long long)&As[cur ^ 1][0][0],
                            &A[(size_t)m0 * K + k1], 32, 128, K, M, K, 3, 3);
                tdm_load_2d((unsigned)(unsigned long long)&Bs[cur ^ 1][0][0],
                            &Bt[(size_t)n0 * K + k1], 32, 256, K, N, K, 3, 3);
                __builtin_amdgcn_s_wait_tensorcnt(2);   // stage k complete
            } else {
                __builtin_amdgcn_s_wait_tensorcnt(0);
            }
        }
        __syncthreads();

        Frag af[4];
#pragma unroll
        for (int mt = 0; mt < 4; ++mt) {
            int row = wm * 64 + mt * 16 + l16;
            af[mt].u[0] = *(const uint4*)&As[cur][row][hl * 8];
            af[mt].u[1] = *(const uint4*)&As[cur][row][hl * 8 + 16];
        }
#pragma unroll
        for (int nt = 0; nt < 4; ++nt) {
            Frag bf;
            int col = wn * 64 + nt * 16 + l16;
            bf.u[0] = *(const uint4*)&Bs[cur][col][hl * 8];
            bf.u[1] = *(const uint4*)&Bs[cur][col][hl * 8 + 16];
#pragma unroll
            for (int mt = 0; mt < 4; ++mt)
                acc[mt][nt] = __builtin_amdgcn_wmma_f32_16x16x32_bf16(
                    false, af[mt].v, false, bf.v, (short)0, acc[mt][nt], false, false);
        }
        __syncthreads();   // everyone done reading buf(cur) before refill
    }

    // epilogue: C-fragment element (row = j + 8*hl, col = l16)
#pragma unroll
    for (int mt = 0; mt < 4; ++mt)
#pragma unroll
        for (int nt = 0; nt < 4; ++nt)
#pragma unroll
            for (int j = 0; j < 8; ++j) {
                int row = m0 + wm * 64 + mt * 16 + j + hl * 8;
                int col = n0 + wn * 64 + nt * 16 + l16;
                float v = acc[mt][nt][j];
                if (OUT_F32) ((float*)C)[(size_t)row * N + col] = v;
                else         ((u16*)C)[(size_t)row * N + col] = f2bf(v);
            }
}

// ---------------------------------------------------------------------------
// Build V^T: vt[b][hk][d][s] from kv_buf
// ---------------------------------------------------------------------------
__global__ __launch_bounds__(256) void transpose_v(const u16* __restrict__ kvb,
                                                   u16* __restrict__ vtb) {
    __shared__ u16 T[64][136];
    const int tid = threadIdx.x;
    const int st = blockIdx.x & 31;
    const int h  = (blockIdx.x >> 5) & 7;
    const int b  = blockIdx.x >> 8;
    const int s0 = st * 64;

#pragma unroll
    for (int p = 0; p < 4; ++p) {
        int sl = (tid >> 4) + p * 16;
        int d8 = (tid & 15) * 8;
        *(uint4*)&T[sl][d8] =
            *(const uint4*)&kvb[(size_t)(b * 2048 + s0 + sl) * 2048 + 1024 + h * 128 + d8];
    }
    __syncthreads();
#pragma unroll
    for (int p = 0; p < 4; ++p) {
        int dl = (tid >> 3) + p * 32;
        int s8 = (tid & 7) * 8;
        unsigned w0 = (unsigned)T[s8 + 0][dl] | ((unsigned)T[s8 + 1][dl] << 16);
        unsigned w1 = (unsigned)T[s8 + 2][dl] | ((unsigned)T[s8 + 3][dl] << 16);
        unsigned w2 = (unsigned)T[s8 + 4][dl] | ((unsigned)T[s8 + 5][dl] << 16);
        unsigned w3 = (unsigned)T[s8 + 6][dl] | ((unsigned)T[s8 + 7][dl] << 16);
        *(uint4*)&vtb[(size_t)((b * 8 + h) * 128 + dl) * 2048 + s0 + s8] =
            make_uint4(w0, w1, w2, w3);
    }
}

// ---------------------------------------------------------------------------
// Causal flash attention, GROUP = 4.
// Block = 8 waves = 128 query rows (16/wave), shared double-buffered K/V
// tiles (64 keys) staged with async global->LDS; block-uniform tile count.
// ---------------------------------------------------------------------------
__global__ __launch_bounds__(256) void attn_fwd(const u16* __restrict__ qb,
                                                const u16* __restrict__ kvb,
                                                const u16* __restrict__ vtb,
                                                u16* __restrict__ ob) {
    const int tid  = threadIdx.x;
    const int wave = tid >> 5;
    const int l    = tid & 31;
    const int l16  = l & 15;
    const int hl   = l >> 4;

    const int qt = blockIdx.x & 15;          // q-tile of 128 rows
    const int h  = (blockIdx.x >> 4) & 31;   // query head
    const int b  = blockIdx.x >> 9;          // batch
    const int hk = h >> 2;                   // kv head

    const int qs0 = qt * 128 + wave * 16;    // wave's first query (seq pos)

    __shared__ u16 Ks[2][64][136];           // [key][dim], +8 pad
    __shared__ u16 Vs[2][128][72];           // [dim][key], +8 pad
    __shared__ u16 P[8][16][72];             // per-wave P tile (16 x 64)

    // Q fragments: 16 rows x 128 dims = 4 K-chunks (persistent in VGPRs)
    Frag qf[4];
    {
        const size_t qrow = (size_t)(b * 2048 + qs0 + l16) * 4096 + h * 128;
#pragma unroll
        for (int kc = 0; kc < 4; ++kc) {
            qf[kc].u[0] = *(const uint4*)&qb[qrow + kc * 32 + hl * 8];
            qf[kc].u[1] = *(const uint4*)&qb[qrow + kc * 32 + hl * 8 + 16];
        }
    }

    v8f o[8] = {};
    float mi[8], li[8];
#pragma unroll
    for (int j = 0; j < 8; ++j) { mi[j] = -1e30f; li[j] = 0.0f; }

    const float scale = 0.08838834764831845f;   // 1/sqrt(128)
    const float L2E   = 1.4426950408889634f;

    const int nkt = (qt * 128 + 128) >> 6;      // block-uniform causal bound

    // stage key-tile 0
    {
#pragma unroll
        for (int i = 0; i < 4; ++i) {            // K: 64 x 128
            int idx = tid + i * 256;
            int kl = idx >> 4, d8 = (idx & 15) * 8;
            async_b128(&kvb[(size_t)(b * 2048 + kl) * 2048 + hk * 128 + d8],
                       &Ks[0][kl][d8]);
        }
#pragma unroll
        for (int i = 0; i < 4; ++i) {            // V^T: 128 x 64
            int idx = tid + i * 256;
            int dl = idx >> 3, s8 = (idx & 7) * 8;
            async_b128(&vtb[(size_t)((b * 8 + hk) * 128 + dl) * 2048 + s8],
                       &Vs[0][dl][s8]);
        }
    }

    for (int kt = 0; kt < nkt; ++kt) {
        const int kb  = kt << 6;
        const int cur = kt & 1;

        if (kt + 1 < nkt) {
            const int kb1 = kb + 64;
#pragma unroll
            for (int i = 0; i < 4; ++i) {
                int idx = tid + i * 256;
                int kl = idx >> 4, d8 = (idx & 15) * 8;
                async_b128(&kvb[(size_t)(b * 2048 + kb1 + kl) * 2048 + hk * 128 + d8],
                           &Ks[cur ^ 1][kl][d8]);
            }
#pragma unroll
            for (int i = 0; i < 4; ++i) {
                int idx = tid + i * 256;
                int dl = idx >> 3, s8 = (idx & 7) * 8;
                async_b128(&vtb[(size_t)((b * 8 + hk) * 128 + dl) * 2048 + kb1 + s8],
                           &Vs[cur ^ 1][dl][s8]);
            }
            asm volatile("s_wait_asynccnt 8" ::: "memory");   // tile kt done
        } else {
            asm volatile("s_wait_asynccnt 0" ::: "memory");
        }
        __syncthreads();

        // -------- S = Q K^T --------
        v8f s[4] = {};
#pragma unroll
        for (int nt = 0; nt < 4; ++nt) {
            const int krow = nt * 16 + l16;
#pragma unroll
            for (int kc = 0; kc < 4; ++kc) {
                Frag kf;
                kf.u[0] = *(const uint4*)&Ks[cur][krow][kc * 32 + hl * 8];
                kf.u[1] = *(const uint4*)&Ks[cur][krow][kc * 32 + hl * 8 + 16];
                s[nt] = __builtin_amdgcn_wmma_f32_16x16x32_bf16(
                    false, qf[kc].v, false, kf.v, (short)0, s[nt], false, false);
            }
        }

        // -------- scale + causal mask + online softmax --------
        float mn[8];
#pragma unroll
        for (int j = 0; j < 8; ++j) mn[j] = mi[j];
#pragma unroll
        for (int nt = 0; nt < 4; ++nt) {
            int key = kb + nt * 16 + l16;
#pragma unroll
            for (int j = 0; j < 8; ++j) {
                int qpos = qs0 + j + hl * 8;
                float v = s[nt][j] * scale;
                v = (key <= qpos) ? v : -1e30f;
                s[nt][j] = v;
                mn[j] = fmaxf(mn[j], v);
            }
        }
#pragma unroll
        for (int j = 0; j < 8; ++j)
#pragma unroll
            for (int d = 1; d < 16; d <<= 1)
                mn[j] = fmaxf(mn[j], __shfl_xor(mn[j], d, 16));

        float al[8], rs[8];
#pragma unroll
        for (int j = 0; j < 8; ++j) {
            al[j] = exp2f((mi[j] - mn[j]) * L2E);
            mi[j] = mn[j];
            rs[j] = 0.0f;
        }
#pragma unroll
        for (int nt = 0; nt < 4; ++nt)
#pragma unroll
            for (int j = 0; j < 8; ++j) {
                float p = exp2f((s[nt][j] - mi[j]) * L2E);
                s[nt][j] = p;
                rs[j] += p;
            }
#pragma unroll
        for (int j = 0; j < 8; ++j) {
#pragma unroll
            for (int d = 1; d < 16; d <<= 1) rs[j] += __shfl_xor(rs[j], d, 16);
            li[j] = li[j] * al[j] + rs[j];
        }
#pragma unroll
        for (int nt2 = 0; nt2 < 8; ++nt2)
#pragma unroll
            for (int j = 0; j < 8; ++j) o[nt2][j] *= al[j];

        // -------- stage P (C layout -> A layout) through per-wave LDS --------
#pragma unroll
        for (int nt = 0; nt < 4; ++nt)
#pragma unroll
            for (int j = 0; j < 8; ++j)
                P[wave][j + hl * 8][nt * 16 + l16] = f2bf(s[nt][j]);
        // same-wave DS ops are in-order: writes visible to following reads.

        // -------- O += P V --------
#pragma unroll
        for (int kc = 0; kc < 2; ++kc) {
            Frag pf;
            pf.u[0] = *(const uint4*)&P[wave][l16][kc * 32 + hl * 8];
            pf.u[1] = *(const uint4*)&P[wave][l16][kc * 32 + hl * 8 + 16];
#pragma unroll
            for (int nt2 = 0; nt2 < 8; ++nt2) {
                Frag vf;
                vf.u[0] = *(const uint4*)&Vs[cur][nt2 * 16 + l16][kc * 32 + hl * 8];
                vf.u[1] = *(const uint4*)&Vs[cur][nt2 * 16 + l16][kc * 32 + hl * 8 + 16];
                o[nt2] = __builtin_amdgcn_wmma_f32_16x16x32_bf16(
                    false, pf.v, false, vf.v, (short)0, o[nt2], false, false);
            }
        }
        __syncthreads();   // done reading buf(cur) before refill
    }

    // -------- normalize + store attn (bf16, row-major [B*S][4096]) --------
#pragma unroll
    for (int nt2 = 0; nt2 < 8; ++nt2)
#pragma unroll
        for (int j = 0; j < 8; ++j) {
            int qpos = qs0 + j + hl * 8;
            float v = o[nt2][j] / li[j];
            ob[(size_t)(b * 2048 + qpos) * 4096 + h * 128 + nt2 * 16 + l16] = f2bf(v);
        }
}

// ---------------------------------------------------------------------------
// Host launch
// ---------------------------------------------------------------------------
extern "C" void kernel_launch(void* const* d_in, const int* in_sizes, int n_in,
                              void* d_out, int out_size, void* d_ws, size_t ws_size,
                              hipStream_t stream) {
    const float* x   = (const float*)d_in[0];   // [2,2048,4096]
    const float* Wq  = (const float*)d_in[1];   // [4096,4096]
    const float* Wkv = (const float*)d_in[2];   // [4096,2048]
    const float* Wo  = (const float*)d_in[3];   // [4096,4096]
    float* out = (float*)d_out;                  // [2,2048,4096] fp32

    char* ws = (char*)d_ws;
    const size_t MB = (size_t)1 << 20;
    u16* x_bf   = (u16*)(ws + 0 * MB);     // 32 MB  [4096][4096]
    u16* wq_t   = (u16*)(ws + 32 * MB);    // 32 MB  [4096][4096]
    u16* wkv_t  = (u16*)(ws + 64 * MB);    // 16 MB  [2048][4096]
    u16* wo_t   = (u16*)(ws + 80 * MB);    // 32 MB  [4096][4096]
    u16* q_buf  = (u16*)(ws + 112 * MB);   // 32 MB  [4096][4096]
    u16* kv_buf = (u16*)(ws + 144 * MB);   // 16 MB  [4096][2048]
    u16* vt_buf = (u16*)(ws + 160 * MB);   //  8 MB  [2][8][128][2048]
    u16* at_buf = (u16*)(ws + 168 * MB);   // 32 MB  [4096][4096]

    cvt_bf16<<<16384, 256, 0, stream>>>((const float4*)x, x_bf, 4194304);
    transpose_cvt<<<dim3(128, 128), 256, 0, stream>>>(Wq, wq_t, 4096, 4096);
    transpose_cvt<<<dim3(64, 128), 256, 0, stream>>>(Wkv, wkv_t, 4096, 2048);
    transpose_cvt<<<dim3(128, 128), 256, 0, stream>>>(Wo, wo_t, 4096, 4096);
    gemm_bf16<false><<<dim3(16, 32), 256, 0, stream>>>(x_bf, wq_t, q_buf, 4096, 4096, 4096);
    gemm_bf16<false><<<dim3(8, 32), 256, 0, stream>>>(x_bf, wkv_t, kv_buf, 4096, 2048, 4096);
    transpose_v<<<512, 256, 0, stream>>>(kv_buf, vt_buf);
    attn_fwd<<<1024, 256, 0, stream>>>(q_buf, kv_buf, vt_buf, at_buf);
    gemm_bf16<true><<<dim3(16, 32), 256, 0, stream>>>(at_buf, wo_t, out, 4096, 4096, 4096);
}
